// Proxy_Anchor_60971355734204
// MI455X (gfx1250) — matrix operand order repaired and checked by reading, not compiled, not run on hardware
//
#include <hip/hip_runtime.h>
#include <hip/hip_bf16.h>
#include <math.h>

typedef __attribute__((ext_vector_type(4)))  __bf16 v4bf;
typedef __attribute__((ext_vector_type(8)))  __bf16 v8bf;
typedef __attribute__((ext_vector_type(16))) __bf16 v16bf;
typedef __attribute__((ext_vector_type(8)))  float  v8f;

#define NB_CLASSES 1024
#define SZ_EMBED   128
#define ROWS       64     // X rows staged per block (4 strips of 16)
#define XST        144    // LDS row stride in bf16: 288B -> 16B aligned, bank-stride 8
#define ALPHA      32.0f
#define MRG        0.1f

// A-fragment (16-bit A 16x32): per lane one contiguous 16-element K-run at +boffA
__device__ __forceinline__ v16bf frag_a(const __bf16* p) {
    v8bf a = *(const v8bf*)p;          // ds_load_b128
    v8bf b = *(const v8bf*)(p + 8);    // ds_load_b128
    return __builtin_shufflevector(a, b, 0,1,2,3,4,5,6,7,8,9,10,11,12,13,14,15);
}

__device__ __forceinline__ v8f wmma_bf16(v16bf A, v16bf B, v8f C) {
    return __builtin_amdgcn_wmma_f32_16x16x32_bf16(
        /*neg_a=*/false, A, /*neg_b=*/false, B,
        /*c_mod=*/(short)0, C, /*reuse_a=*/false, /*reuse_b=*/false);
}

// ---------------------------------------------------------------------------
// Kernel 0: normalize proxies, split into bf16 hi/lo, zero global accumulators.
// 1024 blocks x 32 threads (one wave per proxy row).
// ---------------------------------------------------------------------------
__global__ __launch_bounds__(32) void pa_prep(const float* __restrict__ proxies,
                                              __bf16* __restrict__ proxHi,
                                              __bf16* __restrict__ proxLo,
                                              float* __restrict__ Pg,
                                              float* __restrict__ Ng) {
    const int b    = blockIdx.x;
    const int lane = threadIdx.x;
    const float4 v = ((const float4*)(proxies + (size_t)b * SZ_EMBED))[lane];
    float ss = v.x * v.x + v.y * v.y + v.z * v.z + v.w * v.w;
    #pragma unroll
    for (int off = 16; off > 0; off >>= 1) ss += __shfl_xor(ss, off);
    const float s = rsqrtf(ss + 1e-12f);
    const float n0 = v.x * s, n1 = v.y * s, n2 = v.z * s, n3 = v.w * s;
    v4bf hi, lo;
    hi[0] = (__bf16)n0; lo[0] = (__bf16)(n0 - (float)hi[0]);
    hi[1] = (__bf16)n1; lo[1] = (__bf16)(n1 - (float)hi[1]);
    hi[2] = (__bf16)n2; lo[2] = (__bf16)(n2 - (float)hi[2]);
    hi[3] = (__bf16)n3; lo[3] = (__bf16)(n3 - (float)hi[3]);
    ((v4bf*)(proxHi + (size_t)b * SZ_EMBED))[lane] = hi;
    ((v4bf*)(proxLo + (size_t)b * SZ_EMBED))[lane] = lo;
    if (lane == 0) { Pg[b] = 0.0f; Ng[b] = 0.0f; }
}

// ---------------------------------------------------------------------------
// Kernel 1: split-bf16 GEMM (3x v_wmma_f32_16x16x32_bf16) + exp epilogue.
// 4096 blocks x 64 threads (2 waves); each wave owns 2 strips (32 rows),
// sweeps all 64 class tiles with A register-resident.
// ---------------------------------------------------------------------------
__global__ __launch_bounds__(64) void pa_main(const float* __restrict__ X,
                                              const int*   __restrict__ T,
                                              const __bf16* __restrict__ proxHi,
                                              const __bf16* __restrict__ proxLo,
                                              float* __restrict__ Pg,
                                              float* __restrict__ Ng) {
    __shared__ __bf16 Xhi[ROWS * XST];
    __shared__ __bf16 Xlo[ROWS * XST];
    __shared__ float  Pl[NB_CLASSES];
    __shared__ float  Nl[NB_CLASSES];
    __shared__ int    Tl[ROWS];

    const int tid     = threadIdx.x;
    const int lane    = tid & 31;
    const int wave    = tid >> 5;            // 0..1
    const int rowbase = blockIdx.x * ROWS;

    // ---- stage: normalize 32 rows per wave, split f32 -> bf16 hi/lo --------
    #pragma unroll 1
    for (int r = 0; r < 32; ++r) {
        const int row = wave * 32 + r;
        const float4 v = ((const float4*)(X + (size_t)(rowbase + row) * SZ_EMBED))[lane];
        float ss = v.x * v.x + v.y * v.y + v.z * v.z + v.w * v.w;
        #pragma unroll
        for (int off = 16; off > 0; off >>= 1) ss += __shfl_xor(ss, off);
        const float s = rsqrtf(ss + 1e-12f);
        const float n0 = v.x * s, n1 = v.y * s, n2 = v.z * s, n3 = v.w * s;
        v4bf hi, lo;
        hi[0] = (__bf16)n0; lo[0] = (__bf16)(n0 - (float)hi[0]);
        hi[1] = (__bf16)n1; lo[1] = (__bf16)(n1 - (float)hi[1]);
        hi[2] = (__bf16)n2; lo[2] = (__bf16)(n2 - (float)hi[2]);
        hi[3] = (__bf16)n3; lo[3] = (__bf16)(n3 - (float)hi[3]);
        ((v4bf*)(Xhi + row * XST))[lane] = hi;   // row*288B + lane*8B
        ((v4bf*)(Xlo + row * XST))[lane] = lo;
    }
    if (tid < ROWS) Tl[tid] = T[rowbase + tid];
    for (int i = tid; i < NB_CLASSES; i += 64) { Pl[i] = 0.0f; Nl[i] = 0.0f; }
    __syncthreads();

    // ---- fragment addressing (V_WMMA_F32_16X16X32_BF16 layouts) ------------
    const int m     = lane & 15;
    const int g     = lane >> 4;
    const int boffA = g * 16;                // A: contiguous K-run [boffA,+16)
    const int boffB = g * 16;                // B: contiguous K-run [boffB,+16)
    const int s0    = wave * 2;
    const int s1    = wave * 2 + 1;

    // ---- hoist A fragments into registers (16 x v16bf = 128 VGPRs) ---------
    v16bf A0h[4], A0l[4], A1h[4], A1l[4];
    {
        const __bf16* p0h = Xhi + (s0 * 16 + m) * XST + boffA;
        const __bf16* p0l = Xlo + (s0 * 16 + m) * XST + boffA;
        const __bf16* p1h = Xhi + (s1 * 16 + m) * XST + boffA;
        const __bf16* p1l = Xlo + (s1 * 16 + m) * XST + boffA;
        #pragma unroll
        for (int c = 0; c < 4; ++c) {
            A0h[c] = frag_a(p0h + 32 * c);
            A0l[c] = frag_a(p0l + 32 * c);
            A1h[c] = frag_a(p1h + 32 * c);
            A1l[c] = frag_a(p1l + 32 * c);
        }
    }
    const int radd0 = s0 * 16 + g * 8;       // C layout: lanes>=16 -> M+8
    const int radd1 = s1 * 16 + g * 8;

    // ---- sweep class tiles: B loaded once, feeds 6 WMMAs ------------------
    #pragma unroll 1
    for (int ct = 0; ct < NB_CLASSES / 16; ++ct) {
        const int cls = ct * 16 + m;
        const __bf16* bh = proxHi + (size_t)cls * SZ_EMBED + boffB;
        const __bf16* bl = proxLo + (size_t)cls * SZ_EMBED + boffB;

        v8f a0 = {}, a1 = {};
        #pragma unroll
        for (int c = 0; c < 4; ++c) {
            const v16bf Bh = *(const v16bf*)(bh + 32 * c);  // 32B, L2-hot
            const v16bf Bl = *(const v16bf*)(bl + 32 * c);
            a0 = wmma_bf16(A0l[c], Bh, a0);   // lo*hi
            a0 = wmma_bf16(A0h[c], Bl, a0);   // hi*lo
            a0 = wmma_bf16(A0h[c], Bh, a0);   // hi*hi
            a1 = wmma_bf16(A1l[c], Bh, a1);
            a1 = wmma_bf16(A1h[c], Bl, a1);
            a1 = wmma_bf16(A1h[c], Bh, a1);
        }

        // epilogue: every acc entry of this lane belongs to class `cls`
        float pos = 0.0f, neg = 0.0f;
        #pragma unroll
        for (int r = 0; r < 8; ++r) {
            const float c0 = a0[r];
            if (Tl[radd0 + r] == cls) pos += __expf(-ALPHA * (c0 - MRG));
            else                      neg += __expf( ALPHA * (c0 + MRG));
            const float c1 = a1[r];
            if (Tl[radd1 + r] == cls) pos += __expf(-ALPHA * (c1 - MRG));
            else                      neg += __expf( ALPHA * (c1 + MRG));
        }
        if (pos != 0.0f) atomicAdd(&Pl[cls], pos);   // ds_add_f32
        atomicAdd(&Nl[cls], neg);                    // ds_add_f32
    }
    __syncthreads();

    // ---- flush block partials ----------------------------------------------
    for (int i = tid; i < NB_CLASSES; i += 64) {
        const float p = Pl[i];
        if (p != 0.0f) atomicAdd(&Pg[i], p);
        atomicAdd(&Ng[i], Nl[i]);
    }
}

// ---------------------------------------------------------------------------
// Kernel 2: finalize -> [loss, pos_term, neg_term]
// ---------------------------------------------------------------------------
__global__ __launch_bounds__(256) void pa_final(const float* __restrict__ Pg,
                                                const float* __restrict__ Ng,
                                                float* __restrict__ out) {
    __shared__ float sp[256];
    __shared__ float sn[256];
    __shared__ int   sv[256];
    const int t = threadIdx.x;
    float lp = 0.0f, ln = 0.0f;
    int   nvalid = 0;
    for (int c = t; c < NB_CLASSES; c += 256) {
        const float p = Pg[c];
        const float n = Ng[c];
        if (p > 0.0f) { nvalid++; lp += log1pf(p); }
        ln += log1pf(n);
    }
    sp[t] = lp; sn[t] = ln; sv[t] = nvalid;
    __syncthreads();
    for (int s = 128; s > 0; s >>= 1) {
        if (t < s) { sp[t] += sp[t + s]; sn[t] += sn[t + s]; sv[t] += sv[t + s]; }
        __syncthreads();
    }
    if (t == 0) {
        const float nv  = (float)sv[0];
        const float pos = (nv > 0.0f) ? (sp[0] / fmaxf(nv, 1.0f)) : 0.0f;
        const float neg = sn[0] / (float)NB_CLASSES;
        out[0] = pos + neg;
        out[1] = pos;
        out[2] = neg;
    }
}

// ---------------------------------------------------------------------------
extern "C" void kernel_launch(void* const* d_in, const int* in_sizes, int n_in,
                              void* d_out, int out_size, void* d_ws, size_t ws_size,
                              hipStream_t stream) {
    const float* X       = (const float*)d_in[0];   // [262144, 128]
    const float* proxies = (const float*)d_in[1];   // [1024, 128]
    const int*   T       = (const int*)d_in[2];     // [262144]
    float* out = (float*)d_out;                     // [3]

    __bf16* proxHi = (__bf16*)d_ws;                          // 1024*128 bf16
    __bf16* proxLo = proxHi + NB_CLASSES * SZ_EMBED;         // 1024*128 bf16
    float*  Pg     = (float*)(proxLo + NB_CLASSES * SZ_EMBED);
    float*  Ng     = Pg + NB_CLASSES;

    const int n_samples = in_sizes[2];              // 262144

    pa_prep<<<NB_CLASSES, 32, 0, stream>>>(proxies, proxHi, proxLo, Pg, Ng);
    pa_main<<<n_samples / ROWS, 64, 0, stream>>>(X, T, proxHi, proxLo, Pg, Ng);
    pa_final<<<1, 256, 0, stream>>>(Pg, Ng, out);
}
